// MultiHeadAttention_91199335564042
// MI455X (gfx1250) — matrix operand order repaired and checked by reading, not compiled, not run on hardware
//
#include <hip/hip_runtime.h>
#include <hip/hip_bf16.h>

// ---------------------------------------------------------------------------
// MHA forward for MI455X (gfx1250, wave32, WMMA 16x16x32 bf16).
// cvt fp32->bf16 -> fused QKV GEMM (32x64 tile/wave) + RoPE -> flash
// attention (16 queries x dk=64 per wave, 32-key tiles) -> output projection.
// Whole working set (~70MB) is L2-resident (192MB), so kernels maximize
// WMMA FLOPs per L2 byte: 2 A-frags share each B-frag in the dense GEMMs.
// ---------------------------------------------------------------------------

#define NB   4096      // B*S rows
#define BM   2
#define SM   2048
#define DM   1024
#define HM   16
#define DK   64

typedef __attribute__((ext_vector_type(8)))  float          v8f;
typedef __attribute__((ext_vector_type(8)))  unsigned short v8u;
typedef __attribute__((ext_vector_type(16))) unsigned short v16u;
typedef __attribute__((ext_vector_type(16))) __bf16         bf16x16;

__device__ __forceinline__ unsigned short f2bf(float f) {
  unsigned int u = __float_as_uint(f);
  unsigned int r = u + 0x7FFFu + ((u >> 16) & 1u);   // round-to-nearest-even
  return (unsigned short)(r >> 16);
}

// A-fragment (16x32 bf16, MxK): lane L holds row M=L&15, K = kb..kb+7 and
// kb+16..kb+23 with kb = (L>>4)*8.  Two contiguous 16B loads per lane.
__device__ __forceinline__ v16u load_afrag(const unsigned short* __restrict__ p) {
  v8u lo = *(const v8u*)p;
  v8u hi = *(const v8u*)(p + 16);
  v16u r;
#pragma unroll
  for (int i = 0; i < 8; ++i) { r[i] = lo[i]; r[i + 8] = hi[i]; }
  return r;
}

__device__ __forceinline__ v8f wmma_bf16(v16u a, v16u b, v8f c) {
  return __builtin_amdgcn_wmma_f32_16x16x32_bf16(
      false, __builtin_bit_cast(bf16x16, a),
      false, __builtin_bit_cast(bf16x16, b),
      (short)0, c, false, false);
}

// ---------------------------------------------------------------------------
// fp32 -> bf16 conversion (4 elems / thread)
// ---------------------------------------------------------------------------
__global__ void __launch_bounds__(256) cvt_bf16_kernel(
    const float* __restrict__ in, unsigned short* __restrict__ out, int n) {
  int i = (blockIdx.x * 256 + threadIdx.x) * 4;
  if (i + 3 < n) {
    float4 v = *(const float4*)(in + i);
    unsigned int lo = (unsigned int)f2bf(v.x) | ((unsigned int)f2bf(v.y) << 16);
    unsigned int hi = (unsigned int)f2bf(v.z) | ((unsigned int)f2bf(v.w) << 16);
    uint2 pk; pk.x = lo; pk.y = hi;
    *(uint2*)(out + i) = pk;
  }
}

// ---------------------------------------------------------------------------
// Fused QKV projection + RoPE.
// grid = (N/32, H, 3) ; 1 wave per block, 32x64 output tile (one head,
// two 16-row sub-tiles sharing every weight B-fragment).
// q,k stored [b,h,s,dk] bf16 ; v stored transposed [b,h,dk,s] bf16.
// ---------------------------------------------------------------------------
__global__ void __launch_bounds__(32) qkv_rope_kernel(
    const unsigned short* __restrict__ xb,
    const unsigned short* __restrict__ wqb,
    const unsigned short* __restrict__ wkb,
    const unsigned short* __restrict__ wvb,
    const int* __restrict__ tok,
    unsigned short* __restrict__ qb,
    unsigned short* __restrict__ kb,
    unsigned short* __restrict__ vT) {
  const int lane = threadIdx.x & 31;
  const int ln = lane & 15, half = lane >> 4;
  const int m0 = blockIdx.x * 32;
  const int h = blockIdx.y;
  const int proj = blockIdx.z;
  const int d0 = h * DK;
  const unsigned short* wb = (proj == 0) ? wqb : ((proj == 1) ? wkb : wvb);

  v8f acc[2][4] = {};
  const unsigned short* arow0 = xb + (size_t)(m0 + ln) * DM + half * 8;
  const unsigned short* arow1 = arow0 + (size_t)16 * DM;

  for (int k0 = 0; k0 < DM; k0 += 32) {
    __builtin_prefetch(arow0 + k0 + 128, 0, 3);
    __builtin_prefetch(arow1 + k0 + 128, 0, 3);
    v16u a0 = load_afrag(arow0 + k0);
    v16u a1 = load_afrag(arow1 + k0);
#pragma unroll
    for (int j = 0; j < 4; ++j) {
      const unsigned short* bp =
          wb + (size_t)(d0 + j * 16 + ln) * DM + k0 + half * 8;
      v16u b = load_afrag(bp);
      acc[0][j] = wmma_bf16(a0, b, acc[0][j]);
      acc[1][j] = wmma_bf16(a1, b, acc[1][j]);
    }
  }

  const int bidx = m0 / SM;
  const int s_base = m0 % SM;

  if (proj < 2) {
    unsigned short* outp = (proj == 0) ? qb : kb;
#pragma unroll
    for (int i = 0; i < 2; ++i) {
#pragma unroll
      for (int j = 0; j < 4; ++j) {
        const int n = j * 16 + ln;                         // dk index
        const float freq = __expf(-(float)(n & ~1) * 0.14391156516f); // ln(1e4)/64
        const bool even = (n & 1) == 0;
#pragma unroll
        for (int g = 0; g < 8; ++g) {
          const int s_idx = s_base + i * 16 + half * 8 + g;
          const float pos = (float)tok[bidx * SM + s_idx];
          float sn, cs;
          __sincosf(pos * freq, &sn, &cs);
          float v = acc[i][j][g];
          float p = __shfl_xor(v, 1, 32);                  // paired dk element
          float o = v * cs + sn * (even ? -p : p);
          outp[((size_t)(bidx * HM + h) * SM + s_idx) * DK + n] = f2bf(o);
        }
      }
    }
  } else {
#pragma unroll
    for (int i = 0; i < 2; ++i) {
#pragma unroll
      for (int j = 0; j < 4; ++j) {
        const int n = j * 16 + ln;
#pragma unroll
        for (int g = 0; g < 8; ++g) {
          const int s_idx = s_base + i * 16 + half * 8 + g;
          vT[((size_t)(bidx * HM + h) * DK + n) * SM + s_idx] = f2bf(acc[i][j][g]);
        }
      }
    }
  }
}

// ---------------------------------------------------------------------------
// Flash attention, causal.  grid = (S/16, H, B) ; 1 wave per block.
// 16 query rows x dk=64 output; key tiles of 32 (P tile = one WMMA A operand).
// ---------------------------------------------------------------------------
__global__ void __launch_bounds__(32) attn_kernel(
    const unsigned short* __restrict__ qb,
    const unsigned short* __restrict__ kb,
    const unsigned short* __restrict__ vT,
    unsigned short* __restrict__ ab) {
  __shared__ __align__(16) unsigned short pshm[16 * 32];

  const int lane = threadIdx.x & 31;
  const int ln = lane & 15, half = lane >> 4;
  const int q0 = blockIdx.x * 16;
  const int h = blockIdx.y;
  const int b = blockIdx.z;
  const size_t bh = (size_t)(b * HM + h);

  // Q fragments for K-chunks 0..31 and 32..63
  v16u aq[2];
#pragma unroll
  for (int ks = 0; ks < 2; ++ks)
    aq[ks] = load_afrag(qb + (bh * SM + q0 + ln) * DK + ks * 32 + half * 8);

  float rm[8], rl[8];
  v8f o[4] = {};
#pragma unroll
  for (int g = 0; g < 8; ++g) { rm[g] = -1e30f; rl[g] = 0.0f; }

  for (int kt0 = 0; kt0 <= q0 + 15; kt0 += 32) {
    // S = Q @ K^T  (16x32 scores as two 16x16 f32 frags)
    v8f s[2] = {};
#pragma unroll
    for (int j = 0; j < 2; ++j) {
#pragma unroll
      for (int ks = 0; ks < 2; ++ks) {
        const unsigned short* kp =
            kb + (bh * SM + kt0 + j * 16 + ln) * DK + ks * 32 + half * 8;
        s[j] = wmma_bf16(aq[ks], load_afrag(kp), s[j]);
      }
    }
    // scale + causal mask
#pragma unroll
    for (int j = 0; j < 2; ++j) {
      const int kj = kt0 + j * 16 + ln;
#pragma unroll
      for (int g = 0; g < 8; ++g) {
        const int qi = q0 + half * 8 + g;
        float sv = s[j][g] * 0.125f;                     // 1/sqrt(64)
        s[j][g] = (kj <= qi) ? sv : -1e30f;
      }
    }
    // online softmax per row (rows live within a 16-lane half)
#pragma unroll
    for (int g = 0; g < 8; ++g) {
      float tm = fmaxf(s[0][g], s[1][g]);
#pragma unroll
      for (int off = 8; off >= 1; off >>= 1)
        tm = fmaxf(tm, __shfl_xor(tm, off, 32));
      float mnew = fmaxf(rm[g], tm);
      float sc = __expf(rm[g] - mnew);
      rm[g] = mnew;
      float p0 = __expf(s[0][g] - mnew);
      float p1 = __expf(s[1][g] - mnew);
      float ps = p0 + p1;
#pragma unroll
      for (int off = 8; off >= 1; off >>= 1)
        ps += __shfl_xor(ps, off, 32);
      rl[g] = rl[g] * sc + ps;
#pragma unroll
      for (int j2 = 0; j2 < 4; ++j2) o[j2][g] *= sc;
      const int m = half * 8 + g;
      pshm[m * 32 + ln] = f2bf(p0);                      // C-layout -> LDS
      pshm[m * 32 + 16 + ln] = f2bf(p1);
    }
    // re-read P in A-fragment layout (intra-wave LDS, DS ops in-order)
    v16u pa = load_afrag(&pshm[ln * 32 + half * 8]);
    // O += P @ V   (vT rows are contiguous in key dimension)
#pragma unroll
    for (int j2 = 0; j2 < 4; ++j2) {
      const unsigned short* vp =
          vT + (bh * DK + j2 * 16 + ln) * SM + kt0 + half * 8;
      o[j2] = wmma_bf16(pa, load_afrag(vp), o[j2]);
    }
  }

  // normalize + store to [B,S,D] bf16
#pragma unroll
  for (int g = 0; g < 8; ++g) {
    float inv = 1.0f / rl[g];
    const size_t row = (size_t)b * SM + q0 + half * 8 + g;
#pragma unroll
    for (int j2 = 0; j2 < 4; ++j2)
      ab[row * DM + h * DK + j2 * 16 + ln] = f2bf(o[j2][g] * inv);
  }
}

// ---------------------------------------------------------------------------
// Output projection: out = attn @ wo^T (fp32 output).
// grid = (N/32, D/64) ; 1 wave per block, 32x64 tile (B-frags shared by
// the two 16-row sub-tiles).
// ---------------------------------------------------------------------------
__global__ void __launch_bounds__(32) oproj_kernel(
    const unsigned short* __restrict__ ab,
    const unsigned short* __restrict__ wob,
    float* __restrict__ out) {
  const int lane = threadIdx.x & 31;
  const int ln = lane & 15, half = lane >> 4;
  const int m0 = blockIdx.x * 32;
  const int n0 = blockIdx.y * 64;

  v8f acc[2][4] = {};
  const unsigned short* arow0 = ab + (size_t)(m0 + ln) * DM + half * 8;
  const unsigned short* arow1 = arow0 + (size_t)16 * DM;

  for (int k0 = 0; k0 < DM; k0 += 32) {
    __builtin_prefetch(arow0 + k0 + 128, 0, 3);
    __builtin_prefetch(arow1 + k0 + 128, 0, 3);
    v16u a0 = load_afrag(arow0 + k0);
    v16u a1 = load_afrag(arow1 + k0);
#pragma unroll
    for (int j = 0; j < 4; ++j) {
      const unsigned short* bp =
          wob + (size_t)(n0 + j * 16 + ln) * DM + k0 + half * 8;
      v16u b = load_afrag(bp);
      acc[0][j] = wmma_bf16(a0, b, acc[0][j]);
      acc[1][j] = wmma_bf16(a1, b, acc[1][j]);
    }
  }
#pragma unroll
  for (int i = 0; i < 2; ++i) {
#pragma unroll
    for (int j = 0; j < 4; ++j) {
#pragma unroll
      for (int g = 0; g < 8; ++g)
        out[(size_t)(m0 + i * 16 + half * 8 + g) * DM + n0 + j * 16 + ln] =
            acc[i][j][g];
    }
  }
}

// ---------------------------------------------------------------------------
// Workspace layout (bytes):
//   xb  [N*D bf16]        @ 0         (8 MB)
//   wqb/wkb/wvb/wob       @ 8/10/12/14 MB (2 MB each)
//   qb  [B,H,S,dk bf16]   @ 16 MB     (8 MB)
//   kb                    @ 24 MB     (8 MB)
//   vT  [B,H,dk,S bf16]   @ 32 MB     (8 MB)
//   ab  [B,S,D bf16]      @ 40 MB     (8 MB)   -> total 48 MB
// ---------------------------------------------------------------------------
extern "C" void kernel_launch(void* const* d_in, const int* in_sizes, int n_in,
                              void* d_out, int out_size, void* d_ws, size_t ws_size,
                              hipStream_t stream) {
  const float* x  = (const float*)d_in[0];
  const int*  tok = (const int*)d_in[1];
  const float* wq = (const float*)d_in[2];
  const float* wk = (const float*)d_in[3];
  const float* wv = (const float*)d_in[4];
  const float* wo = (const float*)d_in[5];
  float* out = (float*)d_out;

  char* ws = (char*)d_ws;
  unsigned short* xb  = (unsigned short*)(ws);
  unsigned short* wqb = (unsigned short*)(ws + (8u  << 20));
  unsigned short* wkb = (unsigned short*)(ws + (10u << 20));
  unsigned short* wvb = (unsigned short*)(ws + (12u << 20));
  unsigned short* wob = (unsigned short*)(ws + (14u << 20));
  unsigned short* qb  = (unsigned short*)(ws + (16u << 20));
  unsigned short* kb  = (unsigned short*)(ws + (24u << 20));
  unsigned short* vT  = (unsigned short*)(ws + (32u << 20));
  unsigned short* ab  = (unsigned short*)(ws + (40u << 20));

  const int nX = NB * DM;        // 4,194,304
  const int nW = DM * DM;        // 1,048,576
  cvt_bf16_kernel<<<nX / 1024, 256, 0, stream>>>(x,  xb,  nX);
  cvt_bf16_kernel<<<nW / 1024, 256, 0, stream>>>(wq, wqb, nW);
  cvt_bf16_kernel<<<nW / 1024, 256, 0, stream>>>(wk, wkb, nW);
  cvt_bf16_kernel<<<nW / 1024, 256, 0, stream>>>(wv, wvb, nW);
  cvt_bf16_kernel<<<nW / 1024, 256, 0, stream>>>(wo, wob, nW);

  dim3 gq(NB / 32, HM, 3);
  qkv_rope_kernel<<<gq, 32, 0, stream>>>(xb, wqb, wkb, wvb, tok, qb, kb, vT);

  dim3 ga(SM / 16, HM, BM);
  attn_kernel<<<ga, 32, 0, stream>>>(qb, kb, vT, ab);

  dim3 go(NB / 32, DM / 64);
  oproj_kernel<<<go, 32, 0, stream>>>(ab, wob, out);

  (void)in_sizes; (void)n_in; (void)out_size; (void)ws_size;
}
